// ConsciousAttentionLayer_19026705122057
// MI455X (gfx1250) — matrix operand order repaired and checked by reading, not compiled, not run on hardware
//
#include <hip/hip_runtime.h>

// ---------------- problem constants ----------------
static constexpr int Bc  = 2;
static constexpr int Sc  = 2048;
static constexpr int Hc  = 1024;
static constexpr int NHc = 16;
static constexpr int Mc  = Bc * Sc;          // 4096 token rows
static constexpr float CAUSAL_W = 0.7f;
static constexpr float META_W   = 0.5f * 0.3f;   // mw * 0.3 = 0.15

typedef __attribute__((ext_vector_type(16))) _Float16 v16h;
typedef __attribute__((ext_vector_type(8)))  float    v8f;

#define WMMA_F16(a, b, c) \
  __builtin_amdgcn_wmma_f32_16x16x32_f16(false, (a), false, (b), (short)0, (c), false, false)

__device__ __forceinline__ v16h pack16(uint4 lo, uint4 hi) {
  union { uint4 q[2]; v16h v; } u;
  u.q[0] = lo; u.q[1] = hi;
  return u.v;
}

// output modes (template parameter -> branch-free epilogues)
static constexpr int OUT_F32       = 0;  // out = acc + bias
static constexpr int OUT_F32_BLEND = 1;  // out = alpha*(acc+bias) + beta*res
static constexpr int OUT_F16       = 2;  // f16 row-major
static constexpr int OUT_F16T      = 3;  // f16 transposed [N][M]

// =====================================================================
// Generic GEMM:  A[M,K](f32) @ W[K,N](f32) + bias, f16 WMMA, f32 accum.
// Requires M%64==0, N%64==0, K%32==0 (true for every call here).
// =====================================================================
static constexpr int BM = 64, BN = 64, BK = 32, LDSP = 40; // padded row (halves)

template <int OUT>
__global__ __launch_bounds__(128) void gemm_kernel(
    const float* __restrict__ A, int lda,
    const float* __restrict__ W, int ldb,
    const float* __restrict__ bias,
    void* __restrict__ outp, int ldo,
    float alpha, float beta, const float* __restrict__ res,
    int M, int N, int K)
{
  __shared__ _Float16 As[BM * LDSP];     // As[r][k]
  __shared__ _Float16 Bs[BN * LDSP];     // Bs[n][k]  (W tile transposed)

  const int t        = threadIdx.x;
  const int lane     = t & 31;
  const int wave     = t >> 5;
  const int wm       = wave >> 1, wn = wave & 1;  // 2x2 wave grid
  const int lane16   = lane & 15;
  const int laneHalf = lane >> 4;
  const int rowBase  = blockIdx.y * BM;
  const int colBase  = blockIdx.x * BN;

  v8f acc[2][2] = {};

  for (int k0 = 0; k0 < K; k0 += BK) {
    // prefetch next K-tile into cache while we stage/compute this one
    // (lowers to global_prefetch_b8 on gfx1250)
    if (k0 + BK < K) {
      __builtin_prefetch(A + (size_t)(rowBase + (t >> 1)) * lda + (k0 + BK) + (t & 1) * 16, 0, 0);
      __builtin_prefetch(W + (size_t)(k0 + BK + (t >> 2)) * ldb + colBase + (t & 3) * 16, 0, 0);
    }

    // stage A (64x32 f32 -> f16): 512 float4 / 128 threads
    #pragma unroll
    for (int i = 0; i < 4; ++i) {
      int id = t + i * 128;                 // 0..511
      int r  = id >> 3;                     // 8 float4 per row
      int c4 = id & 7;
      float4 f = *(const float4*)(A + (size_t)(rowBase + r) * lda + k0 + c4 * 4);
      _Float16* d = &As[r * LDSP + c4 * 4];
      d[0] = (_Float16)f.x; d[1] = (_Float16)f.y;
      d[2] = (_Float16)f.z; d[3] = (_Float16)f.w;
    }
    // stage W transposed (32x64 f32 -> Bs[n][k])
    #pragma unroll
    for (int i = 0; i < 4; ++i) {
      int id = t + i * 128;
      int k  = id >> 4;                     // 16 float4 per k-row
      int c4 = id & 15;
      float4 f = *(const float4*)(W + (size_t)(k0 + k) * ldb + colBase + c4 * 4);
      Bs[(c4 * 4 + 0) * LDSP + k] = (_Float16)f.x;
      Bs[(c4 * 4 + 1) * LDSP + k] = (_Float16)f.y;
      Bs[(c4 * 4 + 2) * LDSP + k] = (_Float16)f.z;
      Bs[(c4 * 4 + 3) * LDSP + k] = (_Float16)f.w;
    }
    __syncthreads();

    v16h af[2], bf[2];
    #pragma unroll
    for (int mt = 0; mt < 2; ++mt) {
      const _Float16* p = &As[(wm * 32 + mt * 16 + lane16) * LDSP + laneHalf * 8];
      af[mt] = pack16(*(const uint4*)p, *(const uint4*)(p + 16));
    }
    #pragma unroll
    for (int nt = 0; nt < 2; ++nt) {
      const _Float16* p = &Bs[(wn * 32 + nt * 16 + lane16) * LDSP + laneHalf * 8];
      bf[nt] = pack16(*(const uint4*)p, *(const uint4*)(p + 16));
    }
    #pragma unroll
    for (int mt = 0; mt < 2; ++mt)
      #pragma unroll
      for (int nt = 0; nt < 2; ++nt)
        acc[mt][nt] = WMMA_F16(af[mt], bf[nt], acc[mt][nt]);
    __syncthreads();
  }

  // epilogue.  C layout: vgpr r, lanes0-15 -> row=r, lanes16-31 -> row=8+r; col=lane16
  float*    outF = (float*)outp;
  _Float16* outH = (_Float16*)outp;
  #pragma unroll
  for (int mt = 0; mt < 2; ++mt)
    #pragma unroll
    for (int nt = 0; nt < 2; ++nt) {
      int col = colBase + wn * 32 + nt * 16 + lane16;
      float bv = bias[col];
      #pragma unroll
      for (int r = 0; r < 8; ++r) {
        int row = rowBase + wm * 32 + mt * 16 + laneHalf * 8 + r;
        float v = acc[mt][nt][r] + bv;
        if (OUT == OUT_F16)            outH[(size_t)row * ldo + col] = (_Float16)v;
        else if (OUT == OUT_F16T)      outH[(size_t)col * ldo + row] = (_Float16)v;
        else if (OUT == OUT_F32)       outF[(size_t)row * ldo + col] = v;
        else /* OUT_F32_BLEND */
          outF[(size_t)row * ldo + col] = alpha * v + beta * res[(size_t)row * ldo + col];
      }
    }
}

// =====================================================================
// Flash attention, one wave = 16 queries of one (batch, head).
// Qh,Kh : f16 [M][Htot] (head h at cols h*D..); Vt : f16 [Htot][M].
// mod   : optional f32 [M][NH] per-(row,head) score multiplier.
// =====================================================================
template <int D, bool USEMOD>
__global__ __launch_bounds__(32) void attn_kernel(
    const _Float16* __restrict__ Qh, const _Float16* __restrict__ Kh,
    const _Float16* __restrict__ Vt, const float* __restrict__ mod,
    const int* __restrict__ mask, float* __restrict__ ctx,
    int S, int Htot, int Mtot, float scale)
{
  constexpr int ND = D / 32;   // QK^T contraction fragments
  constexpr int NT = D / 16;   // ctx column tiles
  const int lane     = threadIdx.x;
  const int lane16   = lane & 15;
  const int laneHalf = lane >> 4;
  const int qt = blockIdx.x, h = blockIdx.y, b = blockIdx.z;
  const size_t qRow = (size_t)b * S + qt * 16;
  const int hOff = h * D;

  // Q fragments (A layout): contiguous 16B loads straight from global
  v16h qf[ND];
  #pragma unroll
  for (int d = 0; d < ND; ++d) {
    const _Float16* p = Qh + (qRow + lane16) * Htot + hOff + d * 32 + laneHalf * 8;
    qf[d] = pack16(*(const uint4*)p, *(const uint4*)(p + 16));
  }

  float mrow[8], lrow[8], smod[8];
  v8f acc[NT] = {};
  #pragma unroll
  for (int r = 0; r < 8; ++r) {
    mrow[r] = -3.0e38f; lrow[r] = 0.0f;
    float sc = scale;
    if (USEMOD) sc *= mod[(qRow + laneHalf * 8 + r) * NHc + h];
    smod[r] = sc;
  }

  __shared__ _Float16 Plds[16 * 32];
  const int* bmask = mask + b * S;

  for (int kb = 0; kb < S; kb += 32) {
    // ---- S = Q K^T for 32 keys (two 16x16 tiles) ----
    v8f s0 = {}, s1 = {};
    #pragma unroll
    for (int d = 0; d < ND; ++d) {
      const _Float16* pk0 = Kh + ((size_t)b * S + kb + lane16) * Htot + hOff + d * 32 + laneHalf * 8;
      const _Float16* pk1 = pk0 + (size_t)16 * Htot;
      v16h k0 = pack16(*(const uint4*)pk0, *(const uint4*)(pk0 + 16));
      v16h k1 = pack16(*(const uint4*)pk1, *(const uint4*)(pk1 + 16));
      s0 = WMMA_F16(qf[d], k0, s0);
      s1 = WMMA_F16(qf[d], k1, s1);
    }

    const int mk0 = bmask[kb + lane16];
    const int mk1 = bmask[kb + 16 + lane16];
    float p0[8], p1[8], corr[8];
    #pragma unroll
    for (int r = 0; r < 8; ++r) {
      float a0 = mk0 ? s0[r] * smod[r] : -1.0e30f;
      float a1 = mk1 ? s1[r] * smod[r] : -1.0e30f;
      float tmax = fmaxf(a0, a1);
      tmax = fmaxf(tmax, __shfl_xor(tmax, 1, 32));
      tmax = fmaxf(tmax, __shfl_xor(tmax, 2, 32));
      tmax = fmaxf(tmax, __shfl_xor(tmax, 4, 32));
      tmax = fmaxf(tmax, __shfl_xor(tmax, 8, 32));
      float mnew = fmaxf(mrow[r], tmax);
      corr[r] = __expf(mrow[r] - mnew);
      mrow[r] = mnew;
      float e0 = __expf(a0 - mnew), e1 = __expf(a1 - mnew);
      p0[r] = e0; p1[r] = e1;
      float ssum = e0 + e1;
      ssum += __shfl_xor(ssum, 1, 32);
      ssum += __shfl_xor(ssum, 2, 32);
      ssum += __shfl_xor(ssum, 4, 32);
      ssum += __shfl_xor(ssum, 8, 32);
      lrow[r] = lrow[r] * corr[r] + ssum;
    }
    #pragma unroll
    for (int t2 = 0; t2 < NT; ++t2)
      #pragma unroll
      for (int r = 0; r < 8; ++r) acc[t2][r] *= corr[r];

    // ---- transpose P to A-fragment layout via LDS ----
    __syncthreads();
    #pragma unroll
    for (int r = 0; r < 8; ++r) {
      int q = laneHalf * 8 + r;
      Plds[q * 32 + lane16]      = (_Float16)p0[r];
      Plds[q * 32 + 16 + lane16] = (_Float16)p1[r];
    }
    __syncthreads();
    v16h pf;
    { const _Float16* pp = &Plds[lane16 * 32 + laneHalf * 8];
      pf = pack16(*(const uint4*)pp, *(const uint4*)(pp + 16)); }

    // ---- ctx += P V  (V^T rows are contiguous along keys) ----
    #pragma unroll
    for (int t2 = 0; t2 < NT; ++t2) {
      const _Float16* pv = Vt + (size_t)(hOff + t2 * 16 + lane16) * Mtot + (size_t)b * S + kb + laneHalf * 8;
      v16h vf = pack16(*(const uint4*)pv, *(const uint4*)(pv + 16));
      acc[t2] = WMMA_F16(pf, vf, acc[t2]);
    }
  }

  #pragma unroll
  for (int t2 = 0; t2 < NT; ++t2) {
    int col = hOff + t2 * 16 + lane16;
    #pragma unroll
    for (int r = 0; r < 8; ++r) {
      size_t row = qRow + laneHalf * 8 + r;
      ctx[row * Htot + col] = acc[t2][r] / lrow[r];
    }
  }
}

// =====================================================================
// modulation = sigmoid(hs @ Wg + bg + cv @ Wa + ba)   -> [M, NH]
// =====================================================================
__global__ void gate_kernel(const float* __restrict__ hs,
                            const float* __restrict__ Wg, const float* __restrict__ bg,
                            const float* __restrict__ cv,
                            const float* __restrict__ Wa, const float* __restrict__ ba,
                            float* __restrict__ mod, int Mtot)
{
  int id = blockIdx.x * blockDim.x + threadIdx.x;
  if (id >= Mtot * NHc) return;
  int row = id >> 4, h = id & 15;
  float aw = ba[h];
  #pragma unroll
  for (int i = 0; i < 16; ++i) aw += cv[i] * Wa[i * NHc + h];
  float g = bg[h];
  const float* x = hs + (size_t)row * Hc;
  for (int k = 0; k < Hc; ++k) g += x[k] * Wg[k * NHc + h];
  float z = g + aw;
  mod[id] = 1.0f / (1.0f + __expf(-z));
}

// ---------------------------------------------------------------------
template <int OUT>
static void gemm(hipStream_t s, const float* A, int lda, const float* W, int ldb,
                 const float* bias, void* outp, int ldo,
                 float alpha, float beta, const float* res, int M, int N, int K)
{
  dim3 grid(N / BN, M / BM), block(128);
  gemm_kernel<OUT><<<grid, block, 0, s>>>(A, lda, W, ldb, bias, outp, ldo,
                                          alpha, beta, res, M, N, K);
}

extern "C" void kernel_launch(void* const* d_in, const int* in_sizes, int n_in,
                              void* d_out, int out_size, void* d_ws, size_t ws_size,
                              hipStream_t stream)
{
  const float* hs   = (const float*)d_in[0];
  const int*   mask = (const int*)  d_in[1];
  const float* cv   = (const float*)d_in[2];
  const float* Wq   = (const float*)d_in[3];  const float* bq  = (const float*)d_in[4];
  const float* Wk   = (const float*)d_in[5];  const float* bk  = (const float*)d_in[6];
  const float* Wv   = (const float*)d_in[7];  const float* bv  = (const float*)d_in[8];
  const float* Wg   = (const float*)d_in[9];  const float* bg  = (const float*)d_in[10];
  const float* Wa   = (const float*)d_in[11]; const float* ba  = (const float*)d_in[12];
  const float* caw  = (const float*)d_in[13]; const float* cab = (const float*)d_in[14];
  const float* cow  = (const float*)d_in[15]; const float* cob = (const float*)d_in[16];
  const float* maw  = (const float*)d_in[17]; const float* mab = (const float*)d_in[18];
  const float* mow  = (const float*)d_in[19]; const float* mob = (const float*)d_in[20];
  const float* Wo   = (const float*)d_in[21]; const float* bo  = (const float*)d_in[22];
  float* out = (float*)d_out;

  char* ws = (char*)d_ws;
  size_t off = 0;
  auto alloc = [&](size_t bytes) -> char* {
    char* p = ws + off;
    off += (bytes + 255) & ~(size_t)255;
    return p;
  };
  _Float16* Qh   = (_Float16*)alloc((size_t)Mc * Hc * 2);
  _Float16* Kh   = (_Float16*)alloc((size_t)Mc * Hc * 2);
  _Float16* Vt   = (_Float16*)alloc((size_t)Mc * Hc * 2);   // [H][M]
  float*    modb = (float*)   alloc((size_t)Mc * NHc * 4);
  float*    ctxA = (float*)   alloc((size_t)Mc * Hc * 4);   // main ctx / final blend
  float*    ctxM = (float*)   alloc((size_t)Mc * Hc * 4);   // mixed ctx
  float*    ctxB = (float*)   alloc((size_t)Mc * Hc * 4);   // branch ctx (causal/meta)

  // 1) gate -> modulation
  gate_kernel<<<(Mc * NHc + 255) / 256, 256, 0, stream>>>(hs, Wg, bg, cv, Wa, ba, modb, Mc);

  // 2) main QKV (f16; V transposed)
  gemm<OUT_F16 >(stream, hs, Hc, Wq, Hc, bq, Qh, Hc, 1.f, 0.f, nullptr, Mc, Hc, Hc);
  gemm<OUT_F16 >(stream, hs, Hc, Wk, Hc, bk, Kh, Hc, 1.f, 0.f, nullptr, Mc, Hc, Hc);
  gemm<OUT_F16T>(stream, hs, Hc, Wv, Hc, bv, Vt, Mc, 1.f, 0.f, nullptr, Mc, Hc, Hc);

  // 3) main attention: 16 heads, d=64, modulated scores
  attn_kernel<64, true><<<dim3(Sc / 16, NHc, Bc), 32, 0, stream>>>(
      Qh, Kh, Vt, modb, mask, ctxA, Sc, Hc, Mc, 0.125f /* 1/sqrt(64) */);

  // 4) causal-branch QKV (column slices of ca_in_w [H][3H])
  gemm<OUT_F16 >(stream, hs, Hc, caw + 0 * Hc, 3 * Hc, cab + 0 * Hc, Qh, Hc, 1.f, 0.f, nullptr, Mc, Hc, Hc);
  gemm<OUT_F16 >(stream, hs, Hc, caw + 1 * Hc, 3 * Hc, cab + 1 * Hc, Kh, Hc, 1.f, 0.f, nullptr, Mc, Hc, Hc);
  gemm<OUT_F16T>(stream, hs, Hc, caw + 2 * Hc, 3 * Hc, cab + 2 * Hc, Vt, Mc, 1.f, 0.f, nullptr, Mc, Hc, Hc);

  // 5) causal attention: 4 heads, d=256
  attn_kernel<256, false><<<dim3(Sc / 16, 4, Bc), 32, 0, stream>>>(
      Qh, Kh, Vt, nullptr, mask, ctxB, Sc, Hc, Mc, 0.0625f /* 1/sqrt(256) */);

  // 6) ctxM = 0.7*(ctxB @ ca_out_w + b) + 0.3*ctxA
  gemm<OUT_F32_BLEND>(stream, ctxB, Hc, cow, Hc, cob, ctxM, Hc,
                      CAUSAL_W, 1.f - CAUSAL_W, ctxA, Mc, Hc, Hc);

  // 7) meta QKV from mixed ctx
  gemm<OUT_F16 >(stream, ctxM, Hc, maw + 0 * Hc, 3 * Hc, mab + 0 * Hc, Qh, Hc, 1.f, 0.f, nullptr, Mc, Hc, Hc);
  gemm<OUT_F16 >(stream, ctxM, Hc, maw + 1 * Hc, 3 * Hc, mab + 1 * Hc, Kh, Hc, 1.f, 0.f, nullptr, Mc, Hc, Hc);
  gemm<OUT_F16T>(stream, ctxM, Hc, maw + 2 * Hc, 3 * Hc, mab + 2 * Hc, Vt, Mc, 1.f, 0.f, nullptr, Mc, Hc, Hc);

  // 8) meta attention: 4 heads, d=256
  attn_kernel<256, false><<<dim3(Sc / 16, 4, Bc), 32, 0, stream>>>(
      Qh, Kh, Vt, nullptr, mask, ctxB, Sc, Hc, Mc, 0.0625f);

  // 9) ctxA = 0.15*(ctxB @ ma_out_w + b) + 0.85*ctxM
  gemm<OUT_F32_BLEND>(stream, ctxB, Hc, mow, Hc, mob, ctxA, Hc,
                      META_W, 1.f - META_W, ctxM, Mc, Hc, Hc);

  // 10) out = ctxA @ Wo + bo
  gemm<OUT_F32>(stream, ctxA, Hc, Wo, Hc, bo, out, Hc, 1.f, 0.f, nullptr, Mc, Hc, Hc);
}
